// SharedAttention_5317169512738
// MI455X (gfx1250) — matrix operand order repaired and checked by reading, not compiled, not run on hardware
//
#include <hip/hip_runtime.h>

// ---------------- problem constants ----------------
#define S_   2048
#define D_   1024
#define H_   16
#define HD_  64
#define S2_  4096   // 2*S (concat key length)
#define D2_  2048   // 2*D (concat feature dim)
#define MASK_MIN (-3.4028234663852886e38f)

typedef __attribute__((ext_vector_type(16))) __bf16 v16bf;
typedef __attribute__((ext_vector_type(8)))  float  v8f;

__device__ __forceinline__ __bf16 f2bf(float f) {
  union { float f; unsigned u; } x; x.f = f;
  unsigned r = x.u + 0x7FFFu + ((x.u >> 16) & 1u);   // round-to-nearest-even
  unsigned short hv = (unsigned short)(r >> 16);
  return __builtin_bit_cast(__bf16, hv);
}

// ---- CDNA5 async copy: global -> LDS, 16 bytes per enabled lane (ASYNCcnt) --
__device__ __forceinline__ void async_copy_b128(unsigned lds_addr, const __bf16* g) {
  asm volatile("global_load_async_to_lds_b128 %0, %1, off"
               :: "v"(lds_addr), "v"(g)
               : "memory");
}
__device__ __forceinline__ void wait_async_0() {
  asm volatile("s_wait_asynccnt 0x0" ::: "memory");
}
__device__ __forceinline__ unsigned lds_off(const void* p) {
  return (unsigned)(uintptr_t)p;        // LDS aperture: low 32 bits = LDS offset
}

// A fragment: 16x32 bf16, lane = row (lane&15), half h = lane>>4.
// elements 0..7  -> K = k + 8h + e      (VGPR0..3)
// elements 8..15 -> K = k + 16 + 8h + e (VGPR4..7)
__device__ __forceinline__ v16bf load_a_frag(const __bf16* __restrict__ arow, int k, int h) {
  v16bf f;
  const __bf16* p0 = arow + k + 8 * h;
  const __bf16* p1 = p0 + 16;
#pragma unroll
  for (int e = 0; e < 8; ++e) { f[e] = p0[e]; f[e + 8] = p1[e]; }
  return f;
}

// same fragment but source is f32 (converted to bf16 in-register)
__device__ __forceinline__ v16bf load_a_frag_f32(const float* __restrict__ arow, int k, int h) {
  v16bf f;
  const float* p0 = arow + k + 8 * h;
  const float* p1 = p0 + 16;
#pragma unroll
  for (int e = 0; e < 8; ++e) { f[e] = f2bf(p0[e]); f[e + 8] = f2bf(p1[e]); }
  return f;
}

// B fragment from B^T row-major: lane = column (lane&15), elems 0..7 at k+8h+e, 8..15 at k+16+8h+e.
__device__ __forceinline__ v16bf load_b_frag(const __bf16* __restrict__ btrow, int k, int h) {
  v16bf f;
  const __bf16* p = btrow + k + 8 * h;
#pragma unroll
  for (int e = 0; e < 8; ++e) { f[e] = p[e]; f[e + 8] = p[e + 16]; }
  return f;
}

// ---------------- format conversion kernels ----------------

__global__ void pack_shared_kernel(const float* __restrict__ hid,
                                   const float* __restrict__ kno,
                                   __bf16* __restrict__ sh) {
  size_t i = (size_t)blockIdx.x * blockDim.x + threadIdx.x;   // < S_*D2_
  int s = (int)(i >> 11);
  int c = (int)(i & (D2_ - 1));
  float v = (c < D_) ? hid[(size_t)s * D_ + c] : kno[(size_t)s * D_ + (c - D_)];
  sh[i] = f2bf(v);
}

// Wt[n][k] = bf16(W[k][n])   (W is [K,N] fp32 row-major)
__global__ void wtrans_kernel(const float* __restrict__ W, __bf16* __restrict__ Wt,
                              int K, int N) {
  size_t i = (size_t)blockIdx.x * blockDim.x + threadIdx.x;
  if (i >= (size_t)K * N) return;
  int n = (int)(i / K), k = (int)(i % K);
  Wt[i] = f2bf(W[(size_t)k * N + n]);
}

// ---------------- generic bf16 GEMM: C = A * B (B given as B^T row-major) ----
// block(256)=8 waves -> 128x64 C tile; grid(M/128, N/64).
// BOTH operand tiles (A: 128x32, B: 64x32 bf16) staged in LDS by
// global_load_async_to_lds_b128, double-buffered; last k-stage peeled so the
// steady-state loop has no branches and no register rotation.
// omode 0: f32 row-major out; 1: bf16 row-major out; 2: bf16 transposed out Ot[n][m+ocol]
__global__ void gemm_bf16_kernel(const __bf16* __restrict__ A, int lda, int aoff,
                                 const __bf16* __restrict__ Bt, int ldb,
                                 int Kd,
                                 float* __restrict__ outF, __bf16* __restrict__ outB,
                                 int ldo, int ocol, int omode) {
  __shared__ __align__(16) __bf16 atile[2][128 * 32];  // 2 x 8 KB
  __shared__ __align__(16) __bf16 btile[2][64 * 32];   // 2 x 4 KB

  int t = threadIdx.x;
  int wave = t >> 5;
  int lane = t & 31;
  int r = lane & 15, h = lane >> 4;
  int mblk = blockIdx.x * 128;
  int n0 = blockIdx.y * 64;

  // ---- cooperative staging addresses (per thread) ----
  // B tile: 64 rows x 32 k -> 1 x b128 per thread
  int sbrow = t >> 2;                 // 0..63
  int sbk   = (t & 3) * 8;            // element offset
  const __bf16* bsrc = Bt + (size_t)(n0 + sbrow) * ldb + sbk;
  unsigned bdst[2] = { lds_off(&btile[0][sbrow * 32 + sbk]),
                       lds_off(&btile[1][sbrow * 32 + sbk]) };
  // A tile: 128 rows x 32 k -> 2 x b128 per thread (rows t>>2 and t>>2 + 64)
  int sarow = t >> 2;
  int sak   = (t & 3) * 8;
  const __bf16* asrc0 = A + (size_t)(mblk + sarow) * lda + aoff + sak;
  const __bf16* asrc1 = asrc0 + (size_t)64 * lda;
  unsigned adst0[2] = { lds_off(&atile[0][sarow * 32 + sak]),
                        lds_off(&atile[1][(sarow) * 32 + sak]) };
  unsigned adst1[2] = { lds_off(&atile[0][(sarow + 64) * 32 + sak]),
                        lds_off(&atile[1][(sarow + 64) * 32 + sak]) };

  v8f acc[4];
  v8f zero = {0.f, 0.f, 0.f, 0.f, 0.f, 0.f, 0.f, 0.f};
#pragma unroll
  for (int j = 0; j < 4; ++j) acc[j] = zero;

  int nk = Kd >> 5;
  int a_l = (wave * 16 + r) * 32;     // A-fragment row base (elements) in LDS tile

  // prime stage 0
  async_copy_b128(bdst[0], bsrc);
  async_copy_b128(adst0[0], asrc0);
  async_copy_b128(adst1[0], asrc1);

  for (int ks = 0; ks < nk - 1; ++ks) {
    wait_async_0();
    __syncthreads();
    {
      int kn = (ks + 1) << 5;
      int sel = (ks + 1) & 1;
      async_copy_b128(bdst[sel], bsrc + kn);
      async_copy_b128(adst0[sel], asrc0 + kn);
      async_copy_b128(adst1[sel], asrc1 + kn);
    }
    const __bf16* at = &atile[ks & 1][0];
    const __bf16* bt = &btile[ks & 1][0];
    v16bf af;
    {
      const __bf16* p0 = at + a_l + 8 * h;
      const __bf16* p1 = p0 + 16;
#pragma unroll
      for (int e = 0; e < 8; ++e) { af[e] = p0[e]; af[e + 8] = p1[e]; }
    }
#pragma unroll
    for (int j = 0; j < 4; ++j) {
      v16bf bfr;
      const __bf16* p = bt + (16 * j + r) * 32 + 8 * h;
#pragma unroll
      for (int e = 0; e < 8; ++e) { bfr[e] = p[e]; bfr[e + 8] = p[e + 16]; }
      acc[j] = __builtin_amdgcn_wmma_f32_16x16x32_bf16(
          false, af, false, bfr, (short)0, acc[j], false, false);
    }
  }

  // peeled last stage
  wait_async_0();
  __syncthreads();
  {
    const __bf16* at = &atile[(nk - 1) & 1][0];
    const __bf16* bt = &btile[(nk - 1) & 1][0];
    v16bf af;
    {
      const __bf16* p0 = at + a_l + 8 * h;
      const __bf16* p1 = p0 + 16;
#pragma unroll
      for (int e = 0; e < 8; ++e) { af[e] = p0[e]; af[e + 8] = p1[e]; }
    }
#pragma unroll
    for (int j = 0; j < 4; ++j) {
      v16bf bfr;
      const __bf16* p = bt + (16 * j + r) * 32 + 8 * h;
#pragma unroll
      for (int e = 0; e < 8; ++e) { bfr[e] = p[e]; bfr[e + 8] = p[e + 16]; }
      acc[j] = __builtin_amdgcn_wmma_f32_16x16x32_bf16(
          false, af, false, bfr, (short)0, acc[j], false, false);
    }
  }

  int m0 = mblk + wave * 16;
#pragma unroll
  for (int j = 0; j < 4; ++j) {
#pragma unroll
    for (int i = 0; i < 8; ++i) {
      int m = m0 + 8 * h + i;          // C layout: VGPR i -> M = i + 8*(lane>>4)
      int n = n0 + 16 * j + r;         // N = lane&15
      float v = acc[j][i];
      if (omode == 0)      outF[(size_t)m * ldo + n] = v;
      else if (omode == 1) outB[(size_t)m * ldo + n] = f2bf(v);
      else                 outB[(size_t)n * ldo + (m + ocol)] = f2bf(v);
    }
  }
}

// ---------------- scores: S = (Q_h . K_h^T)/32 with causal mask -------------
// blockIdx.z: bit0 = which (0=know,1=mlp), bits[5:1] = head
__global__ void score_gemm_kernel(const __bf16* __restrict__ Qb,
                                  const __bf16* __restrict__ Kknow,
                                  const __bf16* __restrict__ Kmlp,
                                  float* __restrict__ scores) {
  int head  = blockIdx.z >> 1;
  int which = blockIdx.z & 1;
  const __bf16* Kb = which ? Kmlp : Kknow;

  int wave = threadIdx.x >> 5;
  int lane = threadIdx.x & 31;
  int r = lane & 15, h = lane >> 4;
  int m0 = blockIdx.x * 128 + wave * 16;
  int n0 = blockIdx.y * 64;

  const __bf16* arow = Qb + (size_t)(m0 + r) * D_ + head * HD_;
  const __bf16* brow[4];
#pragma unroll
  for (int j = 0; j < 4; ++j) brow[j] = Kb + (size_t)(n0 + 16 * j + r) * D_ + head * HD_;

  v8f acc[4];
  v8f zero = {0.f, 0.f, 0.f, 0.f, 0.f, 0.f, 0.f, 0.f};
#pragma unroll
  for (int j = 0; j < 4; ++j) acc[j] = zero;

#pragma unroll
  for (int k = 0; k < HD_; k += 32) {
    v16bf af = load_a_frag(arow, k, h);
#pragma unroll
    for (int j = 0; j < 4; ++j) {
      v16bf bfr = load_b_frag(brow[j], k, h);
      acc[j] = __builtin_amdgcn_wmma_f32_16x16x32_bf16(
          false, af, false, bfr, (short)0, acc[j], false, false);
    }
  }

  const float scale = 1.0f / 32.0f;   // 1/sqrt(D), D = 1024
  float* out = scores + (size_t)head * S_ * S2_ + (size_t)which * S_;
#pragma unroll
  for (int j = 0; j < 4; ++j) {
#pragma unroll
    for (int i = 0; i < 8; ++i) {
      int m = m0 + 8 * h + i;
      int n = n0 + 16 * j + r;
      float v = (n > m) ? MASK_MIN : acc[j][i] * scale;
      out[(size_t)m * S2_ + n] = v;
    }
  }
}

// ---------------- softmax over rows of 4096 (in place, f32) ----------------
__global__ void softmax_kernel(float* __restrict__ scores) {
  __shared__ float red[256];
  float* p = scores + (size_t)blockIdx.x * S2_;
  int t = threadIdx.x;

  float v[16];
  float mx = -INFINITY;
#pragma unroll
  for (int e = 0; e < 16; ++e) { v[e] = p[t + 256 * e]; mx = fmaxf(mx, v[e]); }
  red[t] = mx; __syncthreads();
  for (int s = 128; s > 0; s >>= 1) { if (t < s) red[t] = fmaxf(red[t], red[t + s]); __syncthreads(); }
  mx = red[0]; __syncthreads();

  float sum = 0.f;
#pragma unroll
  for (int e = 0; e < 16; ++e) { v[e] = __expf(v[e] - mx); sum += v[e]; }
  red[t] = sum; __syncthreads();
  for (int s = 128; s > 0; s >>= 1) { if (t < s) red[t] += red[t + s]; __syncthreads(); }
  float inv = 1.0f / red[0];

#pragma unroll
  for (int e = 0; e < 16; ++e) p[t + 256 * e] = v[e] * inv;
}

// ---------------- O_h = P_h (f32, cvt->bf16 in-reg) @ V  (V^T rows) --------
// grid(S/128, 1, H). K=4096, manually software-pipelined 2-deep via explicit
// A/B register ping-pong (no copies, no in-loop branches; tail peeled).
__global__ void av_gemm_kernel(const float* __restrict__ P,
                               const __bf16* __restrict__ Vt,
                               __bf16* __restrict__ Obf) {
  int head = blockIdx.z;
  int wave = threadIdx.x >> 5;
  int lane = threadIdx.x & 31;
  int r = lane & 15, h = lane >> 4;
  int m0 = blockIdx.x * 128 + wave * 16;

  const float* arow = P + (size_t)head * S_ * S2_ + (size_t)(m0 + r) * S2_;
  const __bf16* brow[4];
#pragma unroll
  for (int j = 0; j < 4; ++j) brow[j] = Vt + (size_t)(head * HD_ + 16 * j + r) * S2_;

  v8f acc[4];
  v8f zero = {0.f, 0.f, 0.f, 0.f, 0.f, 0.f, 0.f, 0.f};
#pragma unroll
  for (int j = 0; j < 4; ++j) acc[j] = zero;

  v16bf afA = load_a_frag_f32(arow, 0, h);
  v16bf bA[4];
#pragma unroll
  for (int j = 0; j < 4; ++j) bA[j] = load_b_frag(brow[j], 0, h);

  int k = 0;
  for (; k < S2_ - 64; k += 64) {
    v16bf afB = load_a_frag_f32(arow, k + 32, h);
    v16bf bB[4];
#pragma unroll
    for (int j = 0; j < 4; ++j) bB[j] = load_b_frag(brow[j], k + 32, h);
#pragma unroll
    for (int j = 0; j < 4; ++j)
      acc[j] = __builtin_amdgcn_wmma_f32_16x16x32_bf16(
          false, afA, false, bA[j], (short)0, acc[j], false, false);

    afA = load_a_frag_f32(arow, k + 64, h);
#pragma unroll
    for (int j = 0; j < 4; ++j) bA[j] = load_b_frag(brow[j], k + 64, h);
#pragma unroll
    for (int j = 0; j < 4; ++j)
      acc[j] = __builtin_amdgcn_wmma_f32_16x16x32_bf16(
          false, afB, false, bB[j], (short)0, acc[j], false, false);
  }
  // tail: k == S2_-64, two remaining stages
  {
    v16bf afB = load_a_frag_f32(arow, k + 32, h);
    v16bf bB[4];
#pragma unroll
    for (int j = 0; j < 4; ++j) bB[j] = load_b_frag(brow[j], k + 32, h);
#pragma unroll
    for (int j = 0; j < 4; ++j)
      acc[j] = __builtin_amdgcn_wmma_f32_16x16x32_bf16(
          false, afA, false, bA[j], (short)0, acc[j], false, false);
#pragma unroll
    for (int j = 0; j < 4; ++j)
      acc[j] = __builtin_amdgcn_wmma_f32_16x16x32_bf16(
          false, afB, false, bB[j], (short)0, acc[j], false, false);
  }

#pragma unroll
  for (int j = 0; j < 4; ++j) {
#pragma unroll
    for (int i = 0; i < 8; ++i) {
      int m = m0 + 8 * h + i;
      int n = 16 * j + r;
      Obf[(size_t)m * D_ + head * HD_ + n] = f2bf(acc[j][i]);
    }
  }
}

// ---------------- host-side orchestration ----------------
extern "C" void kernel_launch(void* const* d_in, const int* in_sizes, int n_in,
                              void* d_out, int out_size, void* d_ws, size_t ws_size,
                              hipStream_t stream) {
  (void)in_sizes; (void)n_in; (void)out_size; (void)ws_size;

  const float* hid = (const float*)d_in[0];
  const float* kno = (const float*)d_in[1];
  const float* Wq  = (const float*)d_in[2];
  const float* Wkk = (const float*)d_in[3];
  const float* Wkm = (const float*)d_in[4];
  const float* Wvk = (const float*)d_in[5];
  const float* Wvm = (const float*)d_in[6];
  const float* Wo  = (const float*)d_in[7];

  char* ws = (char*)d_ws;
  __bf16* sharedb = (__bf16*)ws; ws += (size_t)S_ * D2_ * 2;   // [S, 2D]
  __bf16* Wqt     = (__bf16*)ws; ws += (size_t)D_ * D2_ * 2;   // [D, 2D]
  __bf16* Wkkt    = (__bf16*)ws; ws += (size_t)D_ * D_ * 2;
  __bf16* Wkmt    = (__bf16*)ws; ws += (size_t)D_ * D_ * 2;
  __bf16* Wvkt    = (__bf16*)ws; ws += (size_t)D_ * D_ * 2;
  __bf16* Wvmt    = (__bf16*)ws; ws += (size_t)D_ * D_ * 2;
  __bf16* Wot     = (__bf16*)ws; ws += (size_t)D_ * D_ * 2;
  __bf16* Qbf     = (__bf16*)ws; ws += (size_t)S_ * D_ * 2;
  __bf16* Kkb     = (__bf16*)ws; ws += (size_t)S_ * D_ * 2;
  __bf16* Kmb     = (__bf16*)ws; ws += (size_t)S_ * D_ * 2;
  __bf16* Vt      = (__bf16*)ws; ws += (size_t)D_ * S2_ * 2;   // [D, 2S] = concat(Vk,Vm)^T
  __bf16* Obf     = (__bf16*)ws; ws += (size_t)S_ * D_ * 2;

  float* out    = (float*)d_out;
  float* scores = out + (size_t)S_ * D_;   // [H, S, 2S] f32 (second tuple output)

  dim3 blk(256);

  // 1. bf16 packing / weight transposes
  pack_shared_kernel<<<(S_ * D2_) / 256, blk, 0, stream>>>(hid, kno, sharedb);
  wtrans_kernel<<<(D_ * D2_ + 255) / 256, blk, 0, stream>>>(Wq,  Wqt,  D2_, D_);
  wtrans_kernel<<<(D_ * D_  + 255) / 256, blk, 0, stream>>>(Wkk, Wkkt, D_,  D_);
  wtrans_kernel<<<(D_ * D_  + 255) / 256, blk, 0, stream>>>(Wkm, Wkmt, D_,  D_);
  wtrans_kernel<<<(D_ * D_  + 255) / 256, blk, 0, stream>>>(Wvk, Wvkt, D_,  D_);
  wtrans_kernel<<<(D_ * D_  + 255) / 256, blk, 0, stream>>>(Wvm, Wvmt, D_,  D_);
  wtrans_kernel<<<(D_ * D_  + 255) / 256, blk, 0, stream>>>(Wo,  Wot,  D_,  D_);

  // 2. projections (bf16 WMMA, f32 accumulate, async-LDS staged A and B)
  dim3 gproj(S_ / 128, D_ / 64);
  gemm_bf16_kernel<<<gproj, blk, 0, stream>>>(sharedb, D2_, 0,  Wqt,  D2_, D2_, nullptr, Qbf, D_,  0,  1);
  gemm_bf16_kernel<<<gproj, blk, 0, stream>>>(sharedb, D2_, D_, Wkkt, D_,  D_,  nullptr, Kkb, D_,  0,  1);
  gemm_bf16_kernel<<<gproj, blk, 0, stream>>>(sharedb, D2_, 0,  Wkmt, D_,  D_,  nullptr, Kmb, D_,  0,  1);
  gemm_bf16_kernel<<<gproj, blk, 0, stream>>>(sharedb, D2_, D_, Wvkt, D_,  D_,  nullptr, Vt,  S2_, 0,  2);
  gemm_bf16_kernel<<<gproj, blk, 0, stream>>>(sharedb, D2_, 0,  Wvmt, D_,  D_,  nullptr, Vt,  S2_, S_, 2);

  // 3. causal-masked scaled scores for both key sets
  dim3 gsc(S_ / 128, S_ / 64, H_ * 2);
  score_gemm_kernel<<<gsc, blk, 0, stream>>>(Qbf, Kkb, Kmb, scores);

  // 4. softmax over concat dimension (4096), in place in d_out
  softmax_kernel<<<H_ * S_, blk, 0, stream>>>(scores);

  // 5. O = P @ concat(V)
  dim3 gav(S_ / 128, 1, H_);
  av_gemm_kernel<<<gav, blk, 0, stream>>>(scores, Vt, Obf);

  // 6. output projection -> d_out[0 : S*D]
  dim3 gout(S_ / 128, D_ / 64);
  gemm_bf16_kernel<<<gout, blk, 0, stream>>>(Obf, D_, 0, Wot, D_, D_, out, nullptr, D_, 0, 0);
}
